// Decoder_64467459113466
// MI455X (gfx1250) — matrix operand order repaired and checked by reading, not compiled, not run on hardware
//
#include <hip/hip_runtime.h>
#include <cstdint>

// ---------------------------------------------------------------------------
// MI455X / gfx1250, wave32, WMMA 16x16x32 f16 (fp32 accumulate).
// Shapes (fixed by reference): B=32 S=512 E=256 H=256 V=512 NH=8 hd=32.
// ---------------------------------------------------------------------------

typedef _Float16 v16h __attribute__((ext_vector_type(16)));
typedef _Float16 v8h  __attribute__((ext_vector_type(8)));
typedef float    v8f  __attribute__((ext_vector_type(8)));
typedef int      v4i  __attribute__((ext_vector_type(4)));
typedef _Float16 HF;

#if defined(__HIP_DEVICE_COMPILE__) && __has_builtin(__builtin_amdgcn_global_load_async_to_lds_b128)
#define HAVE_ASYNC_LDS 1
#else
#define HAVE_ASYNC_LDS 0
#endif

constexpr int Bv = 32, Sv = 512, Ev = 256, Hv = 256, Vout = 512, NHv = 8, HD = 32;
constexpr int MBS = Bv * Sv;            // 16384 token rows
constexpr int G3H = 3 * Hv;             // 768 gate width

__device__ __forceinline__ float sigm(float x) { return 1.0f / (1.0f + __expf(-x)); }

__device__ __forceinline__ v16h cat8(v8h lo, v8h hi)
{
  return __builtin_shufflevector(lo, hi, 0, 1, 2, 3, 4, 5, 6, 7, 8, 9, 10, 11, 12, 13, 14, 15);
}

// ===========================================================================
// Blocked GEMM: D[M,N](col offset, stride ldd) = A[M,K] @ W[N,K]^T + bias
// Block = 256 thr / 8 waves; block tile 256(M) x 64(N); wave tile 32(M) x 64(N)
// Weight tile (64 x 32 k-slice) staged in LDS via async-to-LDS loads and
// shared by all 8 waves; each wave holds a 2x4 grid of 16x16 accumulators.
// ===========================================================================
template <typename T>
__global__ void __launch_bounds__(256) k_gemm_xwt(const HF* __restrict__ A,
                                                  const HF* __restrict__ W,
                                                  const float* __restrict__ bias,
                                                  T* __restrict__ D,
                                                  int M, int N, int K, int ldd, int coloff)
{
  constexpr int WROW = 40;                        // padded LDS row stride (halfs)
  __shared__ __align__(32) HF wtile[64 * WROW];   // 5 KB staged weight tile
  const int lane = threadIdx.x & 31;
  const int wid  = threadIdx.x >> 5;
  const int m    = lane & 15;
  const int hsel = lane >> 4;

  // this thread's 16B staging chunk: W row `wr`, k sub-chunk `wsub`
  const int wr   = threadIdx.x >> 2;              // 0..63
  const int wsub = threadIdx.x & 3;               // 0..3
  HF* ldst = &wtile[wr * WROW + wsub * 8];

  const int nbn  = N >> 6;
  const int nblk = (M >> 8) * nbn;
  for (int blk = blockIdx.x; blk < nblk; blk += gridDim.x) {
    const int bm  = blk / nbn, bn = blk % nbn;
    const int gm0 = bm * 256 + wid * 32;          // wave's first output row
    const int gn0 = bn * 64;                      // block's first output col

    v8f acc[2][4];
#pragma unroll
    for (int i = 0; i < 2; ++i)
#pragma unroll
      for (int j = 0; j < 4; ++j) {
        const float cb = bias[gn0 + j * 16 + m];  // C layout: col = lane&15
#pragma unroll
        for (int r = 0; r < 8; ++r) acc[i][j][r] = cb;
      }

    const HF* ap0  = A + (size_t)(gm0 + m) * K + hsel * 8;  // A frag k-pattern
    const HF* ap1  = ap0 + (size_t)16 * K;
    const HF* wsrc = W + (size_t)(gn0 + wr) * K + wsub * 8;

    for (int k0 = 0; k0 < K; k0 += 32) {
      // ---- stage W[gn0..gn0+63][k0..k0+31] into LDS (ASYNCcnt path) -------
#if HAVE_ASYNC_LDS
      __builtin_amdgcn_global_load_async_to_lds_b128(
          (__attribute__((address_space(1))) v4i*)(wsrc + k0),
          (__attribute__((address_space(3))) v4i*)ldst, 0, 0);
#if __has_builtin(__builtin_amdgcn_s_wait_asynccnt)
      __builtin_amdgcn_s_wait_asynccnt(0);
#else
      asm volatile("s_wait_asynccnt 0x0" ::: "memory");
#endif
#else
      *(v8h*)ldst = *(const v8h*)(wsrc + k0);     // fallback: load + ds_store
#endif
      __syncthreads();

      __builtin_prefetch(ap0 + k0 + 64, 0, 1);    // global_prefetch_b8 on A stream
      v16h a0 = cat8(*(const v8h*)(ap0 + k0), *(const v8h*)(ap0 + k0 + 16));
      v16h a1 = cat8(*(const v8h*)(ap1 + k0), *(const v8h*)(ap1 + k0 + 16));
#pragma unroll
      for (int j = 0; j < 4; ++j) {
        const HF* bp = &wtile[(j * 16 + m) * WROW + hsel * 8];
        v16h b = cat8(*(const v8h*)bp, *(const v8h*)(bp + 16));
        acc[0][j] = __builtin_amdgcn_wmma_f32_16x16x32_f16(false, a0, false, b, (short)0, acc[0][j], false, false);
        acc[1][j] = __builtin_amdgcn_wmma_f32_16x16x32_f16(false, a1, false, b, (short)0, acc[1][j], false, false);
      }
      __syncthreads();                            // before next k-slice overwrite
    }

#pragma unroll
    for (int i = 0; i < 2; ++i)
#pragma unroll
      for (int j = 0; j < 4; ++j)
#pragma unroll
        for (int r = 0; r < 8; ++r)
          D[(size_t)(gm0 + i * 16 + r + 8 * hsel) * ldd + coloff + gn0 + j * 16 + m] = (T)acc[i][j][r];
  }
}

// ===========================================================================
// Flash attention: one wave per (b, head, 16-query tile); hd==32 is exactly
// one WMMA K-step. Streaming softmax stats live in per-wave LDS.
// V is consumed from a d-major transposed copy Vt[b][h][d][s], so every
// B-fragment (K and V) is a single contiguous 32B vector load.
// ===========================================================================
__global__ void __launch_bounds__(256) k_attn(const HF* __restrict__ Q,
                                              const HF* __restrict__ K,
                                              const HF* __restrict__ Vt,
                                              HF* __restrict__ O)
{
  __shared__ float lds[8][16 * 32 + 48];
  const int lane = threadIdx.x & 31;
  const int ws   = threadIdx.x >> 5;
  float* st = &lds[ws][0];   // [16][32] scaled scores
  float* rm = st + 512;      // running row max [16]
  float* rs = st + 528;      // running row sum [16]
  float* al = st + 544;      // rescale alpha / inv-sum [16]

  const int gw = (blockIdx.x << 3) + ws;   // 0..8191
  const int qt = gw & 31;
  const int hh = (gw >> 5) & 7;
  const int bb = gw >> 8;
  const size_t bS = (size_t)bb * Sv;
  const float scale = 0.17677669529663687f;  // 1/sqrt(32)

  const int m = lane & 15, hsel = lane >> 4;

  // Q fragment (A layout)
  const HF* qp = Q + (bS + qt * 16 + m) * Ev + hh * HD + hsel * 16;
  v16h aq = cat8(*(const v8h*)qp, *(const v8h*)(qp + 8));

  // V-transpose base for this (b, h): lane's column d = m (bv0) / m+16 (bv1)
  const HF* vtb = Vt + ((size_t)(bb * NHv + hh) * HD + m) * Sv + hsel * 16;

  if (lane < 16) { rm[lane] = -3.0e38f; rs[lane] = 0.0f; }
  v8f c0 = {}, c1 = {};
  const v8f zero = {};

  for (int kt = 0; kt < 32; kt += 2) {       // 32 keys per iteration
    // K fragments (B layout): lane = key within tile, contiguous over d
    const HF* kp0 = K + (bS + kt * 16 + m) * Ev + hh * HD + hsel * 16;
    v16h bk0 = *(const v16h*)kp0;
    v16h bk1 = *(const v16h*)(kp0 + (size_t)16 * Ev);

    v8f s0 = __builtin_amdgcn_wmma_f32_16x16x32_f16(false, aq, false, bk0, (short)0, zero, false, false);
    v8f s1 = __builtin_amdgcn_wmma_f32_16x16x32_f16(false, aq, false, bk1, (short)0, zero, false, false);

#pragma unroll
    for (int r = 0; r < 8; ++r) {            // C layout -> LDS row-major
      int row = r + 8 * hsel;
      st[row * 32 + m]      = s0[r] * scale;
      st[row * 32 + 16 + m] = s1[r] * scale;
    }
    __syncthreads();
    if (lane < 16) {                         // per-row streaming softmax stats
      float om = rm[lane], nm = om;
      for (int j = 0; j < 32; ++j) nm = fmaxf(nm, st[lane * 32 + j]);
      float a = __expf(om - nm), sum = 0.0f;
      for (int j = 0; j < 32; ++j) sum += __expf(st[lane * 32 + j] - nm);
      rs[lane] = rs[lane] * a + sum;
      rm[lane] = nm;
      al[lane] = a;
    }
    __syncthreads();
#pragma unroll
    for (int r = 0; r < 8; ++r) {            // rescale accumulated O
      float ar = al[r + 8 * hsel];
      c0[r] *= ar; c1[r] *= ar;
    }
    // P fragment (A layout) from LDS
    float nm = rm[m];
    v16h p;
#pragma unroll
    for (int i = 0; i < 8; ++i) {
      p[i]     = (HF)__expf(st[m * 32 + hsel * 16 + i]     - nm);
      p[i + 8] = (HF)__expf(st[m * 32 + hsel * 16 + 8 + i] - nm);
    }
    // V fragments (B layout) from d-major Vt: contiguous over keys
    v16h bv0 = *(const v16h*)(vtb + kt * 16);
    v16h bv1 = *(const v16h*)(vtb + (size_t)16 * Sv + kt * 16);
    c0 = __builtin_amdgcn_wmma_f32_16x16x32_f16(false, p, false, bv0, (short)0, c0, false, false);
    c1 = __builtin_amdgcn_wmma_f32_16x16x32_f16(false, p, false, bv1, (short)0, c1, false, false);
    __syncthreads();                         // st reused next iteration
  }
  if (lane < 16) al[lane] = 1.0f / rs[lane];
  __syncthreads();
#pragma unroll
  for (int r = 0; r < 8; ++r) {
    int row = r + 8 * hsel;
    float inv = al[row];
    HF* op = O + (bS + qt * 16 + row) * Ev + hh * HD + m;
    op[0]  = (HF)(c0[r] * inv);
    op[16] = (HF)(c1[r] * inv);
  }
}

// ---- V transpose: Vt[b][h][d][s] = V[b*S+s][h*32+d] -------------------------
__global__ void k_vt(const HF* __restrict__ V, HF* __restrict__ Vt)
{
  int i  = blockIdx.x * 256 + threadIdx.x;   // exactly B*NH*HD*S threads
  int s  = i & 511;
  int d  = (i >> 9) & 31;
  int hh = (i >> 14) & 7;
  int bb = i >> 17;
  Vt[i] = V[((size_t)bb * Sv + s) * Ev + hh * HD + d];
}

// ---- single 16x16 tile GEMM for the recurrent (K=256) matmuls ---------------
__device__ __forceinline__ v8f gru_tile(const HF* __restrict__ A,
                                        const HF* __restrict__ W,
                                        const float* __restrict__ bias,
                                        int tm, int tn, int lane)
{
  const int m = lane & 15, hsel = lane >> 4;
  v8f acc;
  const float cb = bias[tn * 16 + m];
#pragma unroll
  for (int r = 0; r < 8; ++r) acc[r] = cb;
  const HF* ap = A + (size_t)(tm * 16 + m) * Hv + hsel * 16;
  const HF* wp = W + (size_t)(tn * 16 + m) * Hv + hsel * 16;
#pragma unroll
  for (int k0 = 0; k0 < Hv; k0 += 32) {
    v16h a = cat8(*(const v8h*)(ap + k0), *(const v8h*)(ap + k0 + 8));
    v16h b = *(const v16h*)(wp + k0);
    acc = __builtin_amdgcn_wmma_f32_16x16x32_f16(false, a, false, b, (short)0, acc, false, false);
  }
  return acc;
}

__device__ __forceinline__ void store_tile_f32(float* __restrict__ D, int ldd,
                                               int tm, int tn, const v8f& acc, int lane)
{
  const int n = lane & 15, hsel = lane >> 4;
#pragma unroll
  for (int r = 0; r < 8; ++r)
    D[(size_t)(tm * 16 + r + 8 * hsel) * ldd + tn * 16 + n] = acc[r];
}

// ---- one GRU time step: both layers + gates, single block -------------------
__global__ void __launch_bounds__(256) k_gru_step(
    const float* __restrict__ gi0,
    const HF* __restrict__ w_hh0, const float* __restrict__ b_hh0,
    const HF* __restrict__ w_ih1, const float* __restrict__ b_ih1,
    const HF* __restrict__ w_hh1, const float* __restrict__ b_hh1,
    HF* __restrict__ h0, HF* __restrict__ h1,
    float* __restrict__ gh0, float* __restrict__ gi1, float* __restrict__ gh1,
    HF* __restrict__ ys, int t)
{
  const int lane = threadIdx.x & 31;
  const int wid  = threadIdx.x >> 5;

  // phase A: gh0[32,768] = h0 @ w_hh0^T + b_hh0   (96 tiles, 8 waves)
  for (int tile = wid; tile < 96; tile += 8) {
    int tm = tile / 48, tn = tile % 48;
    v8f acc = gru_tile(h0, w_hh0, b_hh0, tm, tn, lane);
    store_tile_f32(gh0, G3H, tm, tn, acc, lane);
  }
  __threadfence_block(); __syncthreads();

  // gate 0 (torch order r,z,n): h0 <- (1-z)*n + z*h0
  for (int idx = threadIdx.x; idx < Bv * Hv; idx += 256) {
    int bb = idx >> 8, j = idx & 255;
    const float* gi = gi0 + ((size_t)bb * Sv + t) * G3H;
    const float* gh = gh0 + bb * G3H;
    float r = sigm(gi[j] + gh[j]);
    float z = sigm(gi[Hv + j] + gh[Hv + j]);
    float n = tanhf(gi[2 * Hv + j] + r * gh[2 * Hv + j]);
    float hp = (float)h0[idx];
    h0[idx] = (HF)((1.0f - z) * n + z * hp);
  }
  __threadfence_block(); __syncthreads();

  // phase B: gi1 = h0 @ w_ih1^T + b_ih1 ; gh1 = h1 @ w_hh1^T + b_hh1
  for (int tile = wid; tile < 192; tile += 8) {
    int tt = tile;
    const HF* A; const HF* W; const float* bias; float* D;
    if (tt < 96) { A = h0; W = w_ih1; bias = b_ih1; D = gi1; }
    else         { tt -= 96; A = h1; W = w_hh1; bias = b_hh1; D = gh1; }
    int tm = tt / 48, tn = tt % 48;
    v8f acc = gru_tile(A, W, bias, tm, tn, lane);
    store_tile_f32(D, G3H, tm, tn, acc, lane);
  }
  __threadfence_block(); __syncthreads();

  // gate 1: h1 update + write ys[b, t, :]
  for (int idx = threadIdx.x; idx < Bv * Hv; idx += 256) {
    int bb = idx >> 8, j = idx & 255;
    const float* gi = gi1 + bb * G3H;
    const float* gh = gh1 + bb * G3H;
    float r = sigm(gi[j] + gh[j]);
    float z = sigm(gi[Hv + j] + gh[Hv + j]);
    float n = tanhf(gi[2 * Hv + j] + r * gh[2 * Hv + j]);
    float hp = (float)h1[idx];
    float hn = (1.0f - z) * n + z * hp;
    h1[idx] = (HF)hn;
    ys[((size_t)bb * Sv + t) * Hv + j] = (HF)hn;
  }
}

// ---- small helpers ----------------------------------------------------------
__global__ void k_cvt(const float* __restrict__ s, HF* __restrict__ d, int n)
{
  for (int i = blockIdx.x * blockDim.x + threadIdx.x; i < n; i += gridDim.x * blockDim.x)
    d[i] = (HF)s[i];
}

__global__ void k_embed(const float* __restrict__ tab, const int* __restrict__ idx,
                        HF* __restrict__ out)
{
  int i = blockIdx.x * 256 + threadIdx.x;      // exactly MBS*Ev threads
  int pos = i >> 8;
  int e   = i & 255;
  int id  = idx[pos];
  out[i] = (id != 0) ? (HF)tab[(size_t)id * Ev + e] : (HF)0.0f;
}

__global__ void k_zero(HF* __restrict__ a, int n)
{
  for (int i = blockIdx.x * blockDim.x + threadIdx.x; i < n; i += gridDim.x * blockDim.x)
    a[i] = (HF)0.0f;
}

// ---------------------------------------------------------------------------
extern "C" void kernel_launch(void* const* d_in, const int* in_sizes, int n_in,
                              void* d_out, int out_size, void* d_ws, size_t ws_size,
                              hipStream_t stream)
{
  (void)in_sizes; (void)n_in; (void)out_size; (void)ws_size;
  const int*   an_lo  = (const int*)d_in[0];
  const int*   an_hi  = (const int*)d_in[1];
  const float* emb_lo = (const float*)d_in[2];
  const float* emb_hi = (const float*)d_in[3];
  const float* wq_l = (const float*)d_in[4];  const float* bq_l = (const float*)d_in[5];
  const float* wk_l = (const float*)d_in[6];  const float* bk_l = (const float*)d_in[7];
  const float* wv_l = (const float*)d_in[8];  const float* bv_l = (const float*)d_in[9];
  const float* wo_l = (const float*)d_in[10]; const float* bo_l = (const float*)d_in[11];
  const float* wq_h = (const float*)d_in[12]; const float* bq_h = (const float*)d_in[13];
  const float* wk_h = (const float*)d_in[14]; const float* bk_h = (const float*)d_in[15];
  const float* wv_h = (const float*)d_in[16]; const float* bv_h = (const float*)d_in[17];
  const float* wo_h = (const float*)d_in[18]; const float* bo_h = (const float*)d_in[19];
  const float* w_ih0 = (const float*)d_in[20]; const float* w_hh0 = (const float*)d_in[21];
  const float* b_ih0 = (const float*)d_in[22]; const float* b_hh0 = (const float*)d_in[23];
  const float* w_ih1 = (const float*)d_in[24]; const float* w_hh1 = (const float*)d_in[25];
  const float* b_ih1 = (const float*)d_in[26]; const float* b_hh1 = (const float*)d_in[27];
  const float* fc_w  = (const float*)d_in[28]; const float* fc_b  = (const float*)d_in[29];

  char* wsp = (char*)d_ws;
  auto alloc = [&](size_t bytes) -> char* {
    char* r = wsp; wsp += (bytes + 255) & ~(size_t)255; return r;
  };

  // fp16 weight copies
  HF* wq16l = (HF*)alloc((size_t)Ev * Ev * 2);  HF* wk16l = (HF*)alloc((size_t)Ev * Ev * 2);
  HF* wv16l = (HF*)alloc((size_t)Ev * Ev * 2);  HF* wo16l = (HF*)alloc((size_t)Ev * Ev * 2);
  HF* wq16h = (HF*)alloc((size_t)Ev * Ev * 2);  HF* wk16h = (HF*)alloc((size_t)Ev * Ev * 2);
  HF* wv16h = (HF*)alloc((size_t)Ev * Ev * 2);  HF* wo16h = (HF*)alloc((size_t)Ev * Ev * 2);
  HF* wih0_16 = (HF*)alloc((size_t)G3H * 512 * 2);
  HF* whh0_16 = (HF*)alloc((size_t)G3H * Hv * 2);
  HF* wih1_16 = (HF*)alloc((size_t)G3H * Hv * 2);
  HF* whh1_16 = (HF*)alloc((size_t)G3H * Hv * 2);
  HF* fc16    = (HF*)alloc((size_t)Vout * Hv * 2);
  // activations
  HF* eL  = (HF*)alloc((size_t)MBS * Ev * 2);
  HF* eH  = (HF*)alloc((size_t)MBS * Ev * 2);
  HF* Qb  = (HF*)alloc((size_t)MBS * Ev * 2);
  HF* Kb  = (HF*)alloc((size_t)MBS * Ev * 2);
  HF* Vb  = (HF*)alloc((size_t)MBS * Ev * 2);
  HF* Vt  = (HF*)alloc((size_t)MBS * Ev * 2);   // d-major transposed V
  HF* SO  = (HF*)alloc((size_t)MBS * Ev * 2);
  HF* RNN = (HF*)alloc((size_t)MBS * 512 * 2);
  float* GI0 = (float*)alloc((size_t)MBS * G3H * 4);
  float* GH0 = (float*)alloc((size_t)Bv * G3H * 4);
  float* GI1 = (float*)alloc((size_t)Bv * G3H * 4);
  float* GH1 = (float*)alloc((size_t)Bv * G3H * 4);
  HF* h0p = (HF*)alloc((size_t)Bv * Hv * 2);
  HF* h1p = (HF*)alloc((size_t)Bv * Hv * 2);
  HF* YS  = (HF*)alloc((size_t)MBS * Hv * 2);

  auto cvt = [&](const float* s, HF* d, int n) {
    k_cvt<<<(n + 255) / 256, 256, 0, stream>>>(s, d, n);
  };
  cvt(wq_l, wq16l, Ev * Ev); cvt(wk_l, wk16l, Ev * Ev);
  cvt(wv_l, wv16l, Ev * Ev); cvt(wo_l, wo16l, Ev * Ev);
  cvt(wq_h, wq16h, Ev * Ev); cvt(wk_h, wk16h, Ev * Ev);
  cvt(wv_h, wv16h, Ev * Ev); cvt(wo_h, wo16h, Ev * Ev);
  cvt(w_ih0, wih0_16, G3H * 512); cvt(w_hh0, whh0_16, G3H * Hv);
  cvt(w_ih1, wih1_16, G3H * Hv);  cvt(w_hh1, whh1_16, G3H * Hv);
  cvt(fc_w, fc16, Vout * Hv);

  k_zero<<<32, 256, 0, stream>>>(h0p, Bv * Hv);
  k_zero<<<32, 256, 0, stream>>>(h1p, Bv * Hv);

  k_embed<<<MBS * Ev / 256, 256, 0, stream>>>(emb_lo, an_lo, eL);
  k_embed<<<MBS * Ev / 256, 256, 0, stream>>>(emb_hi, an_hi, eH);

  // block tiles: 256 rows x 64 cols
  const int blkE  = (MBS / 256) * (Ev / 64);    // 256 blocks for [16384x256] GEMMs
  const int blkGI = (MBS / 256) * (G3H / 64);   // 768 blocks
  const int blkFC = (MBS / 256) * (Vout / 64);  // 512 blocks
  const int blkAT = Bv * NHv * (Sv / 16) / 8;   // 1024 blocks, 1 wave per q-tile
  const int blkVT = MBS * Ev / 256;             // transpose grid

  // low-res MHA -> rnn_in[:, :, 0:256]
  k_gemm_xwt<HF><<<blkE, 256, 0, stream>>>(eL, wq16l, bq_l, Qb, MBS, Ev, Ev, Ev, 0);
  k_gemm_xwt<HF><<<blkE, 256, 0, stream>>>(eL, wk16l, bk_l, Kb, MBS, Ev, Ev, Ev, 0);
  k_gemm_xwt<HF><<<blkE, 256, 0, stream>>>(eL, wv16l, bv_l, Vb, MBS, Ev, Ev, Ev, 0);
  k_vt<<<blkVT, 256, 0, stream>>>(Vb, Vt);
  k_attn<<<blkAT, 256, 0, stream>>>(Qb, Kb, Vt, SO);
  k_gemm_xwt<HF><<<blkE, 256, 0, stream>>>(SO, wo16l, bo_l, RNN, MBS, Ev, Ev, 512, 0);
  // high-res MHA -> rnn_in[:, :, 256:512]
  k_gemm_xwt<HF><<<blkE, 256, 0, stream>>>(eH, wq16h, bq_h, Qb, MBS, Ev, Ev, Ev, 0);
  k_gemm_xwt<HF><<<blkE, 256, 0, stream>>>(eH, wk16h, bk_h, Kb, MBS, Ev, Ev, Ev, 0);
  k_gemm_xwt<HF><<<blkE, 256, 0, stream>>>(eH, wv16h, bv_h, Vb, MBS, Ev, Ev, Ev, 0);
  k_vt<<<blkVT, 256, 0, stream>>>(Vb, Vt);
  k_attn<<<blkAT, 256, 0, stream>>>(Qb, Kb, Vt, SO);
  k_gemm_xwt<HF><<<blkE, 256, 0, stream>>>(SO, wo16h, bo_h, RNN, MBS, Ev, Ev, 512, 256);

  // hoisted GRU input projection: GI0[16384,768] = RNN @ w_ih0^T + b_ih0
  k_gemm_xwt<float><<<blkGI, 256, 0, stream>>>(RNN, wih0_16, b_ih0, GI0, MBS, G3H, 512, G3H, 0);

  // sequential scan: only recurrent GEMMs + gates per step
  for (int t = 0; t < Sv; ++t)
    k_gru_step<<<1, 256, 0, stream>>>(GI0, whh0_16, b_hh0, wih1_16, b_ih1,
                                      whh1_16, b_hh1, h0p, h1p, GH0, GI1, GH1, YS, t);

  // FC: d_out[16384,512] = YS @ fc_w^T + fc_b
  k_gemm_xwt<float><<<blkFC, 256, 0, stream>>>(YS, fc16, fc_b, (float*)d_out, MBS, Vout, Hv, Vout, 0);
}
